// MatchLSTM_58411555225684
// MI455X (gfx1250) — matrix operand order repaired and checked by reading, not compiled, not run on hardware
//
#include <hip/hip_runtime.h>

// MI455X (gfx1250) Match-LSTM. wave32, WMMA f16->f32, pipelined async-to-LDS.
typedef __attribute__((ext_vector_type(16))) _Float16 v16h;
typedef __attribute__((ext_vector_type(8)))  float    v8f;
typedef _Float16 f16;

static constexpr int TPn = 600, TQN = 60, Bn = 64;
static constexpr int EMB = 344, EMBP = 352;   // pad K to multiple of 32

// ---------------------------------------------------------------- WMMA helpers
__device__ __forceinline__ int lane_id() { return threadIdx.x & 31; }

// A fragment: 16x32 f16, row-major source with leading dim ld.
__device__ __forceinline__ v16h load_A(const f16* base, int ld) {
  int l = lane_id();
  const f16* p = base + (size_t)(l & 15) * ld + ((l >> 4) ? 16 : 0);
  v16h a;
#pragma unroll
  for (int i = 0; i < 8; ++i) a[i] = p[i];
  p += 8;
#pragma unroll
  for (int i = 0; i < 8; ++i) a[8 + i] = p[i];
  return a;
}

// B fragment: 32x16 where B[k][n] = W[n][k], W row-major [N,K].
__device__ __forceinline__ v16h load_Bt(const f16* W, int ld, int n0, int k0) {
  int l = lane_id();
  const f16* p = W + (size_t)(n0 + (l & 15)) * ld + k0 + ((l >> 4) ? 16 : 0);
  v16h b;
#pragma unroll
  for (int i = 0; i < 16; ++i) b[i] = p[i];
  return b;
}

// C/D fragment: 16x16 f32.
__device__ __forceinline__ v8f load_C(const float* src, int ld) {
  int l = lane_id();
  const float* p = src + (size_t)((l >> 4) ? 8 : 0) * ld + (l & 15);
  v8f c;
#pragma unroll
  for (int v = 0; v < 8; ++v) c[v] = p[(size_t)v * ld];
  return c;
}
__device__ __forceinline__ void store_C(float* dst, int ld, v8f c) {
  int l = lane_id();
  float* p = dst + (size_t)((l >> 4) ? 8 : 0) * ld + (l & 15);
#pragma unroll
  for (int v = 0; v < 8; ++v) p[(size_t)v * ld] = c[v];
}

__device__ __forceinline__ v8f wmma16(v16h a, v16h b, v8f c) {
  return __builtin_amdgcn_wmma_f32_16x16x32_f16(false, a, false, b,
                                                (short)0, c, false, false);
}

__device__ __forceinline__ float fsig(float x) {
  x = fminf(fmaxf(x, -30.f), 30.f);
  return 1.0f / (1.0f + __expf(-x));
}
__device__ __forceinline__ float ftanh(float x) {
  x = fminf(fmaxf(x, -15.f), 15.f);
  float e = __expf(2.0f * x);
  return (e - 1.0f) / (e + 1.0f);
}

// ---------------------------------------------------------------- f32 -> f16 (+K pad)
__global__ void k_cvt_pad(const float* __restrict__ in, f16* __restrict__ out,
                          long rows, int K, int KP) {
  long i = (long)blockIdx.x * blockDim.x + threadIdx.x;
  if (i >= rows * (long)KP) return;
  int kp = (int)(i % KP);
  long r = i / KP;
  out[i] = (kp < K) ? (f16)in[r * K + kp] : (f16)0.0f;
}

// ---------------------------------------------------------------- Y = X @ W^T (+bias)
// X: [M,K] f16, W: [N,K] f16, Y: [M,N] f32. Each wave: one 16x64 strip.
// A tiles are staged through LDS with the gfx1250 async copy engine
// (GLOBAL_LOAD_ASYNC_TO_LDS_B128 / ASYNCcnt), double-buffered and prefetched
// one K-chunk ahead so the DMA overlaps the WMMAs of the current chunk.
__device__ __forceinline__ void stage_tile_async(const f16* gsrc, int K,
                                                 unsigned lbase, int l) {
  // 64 chunks of 16B cover the 1KB (16x32 f16) tile; each lane copies 2 chunks.
#pragma unroll
  for (int i = 0; i < 2; ++i) {
    int q = l + 32 * i;
    const f16* g = gsrc + (size_t)(q >> 2) * K + (q & 3) * 8;
    unsigned loff = lbase + q * 16;
    asm volatile("global_load_async_to_lds_b128 %0, %1, off"
                 :: "v"(loff), "v"(g) : "memory");
  }
}

__global__ void k_gemm(const f16* __restrict__ X, const f16* __restrict__ W,
                       const float* __restrict__ bias, float* __restrict__ Y,
                       int M, int N, int K) {
  extern __shared__ char smem[];
  int l = lane_id();
  int wvl = threadIdx.x >> 5;
  f16* atile0 = (f16*)smem + wvl * 1024;  // 2 buffers x 512 f16 (16x32 tile)
  unsigned base0 = (unsigned)((char*)atile0 - (char*)smem);

  int wave = (int)((blockIdx.x * blockDim.x + threadIdx.x) >> 5);
  int strips = N >> 6;
  int m0 = (wave / strips) << 4;
  int n0 = (wave % strips) << 6;
  if (m0 >= M) return;
  const f16* Xrow = X + (size_t)m0 * K;
  v8f acc[4] = {};
  // prologue: stage tile 0 into buffer 0
  stage_tile_async(Xrow, K, base0, l);
  int kb = 0;
  for (int k0 = 0; k0 < K; k0 += 32, kb ^= 1) {
    if (k0 + 32 < K) {
      // prefetch next tile into the other buffer, then wait with slack:
      // allow the 2 just-issued copies to remain in flight.
      stage_tile_async(Xrow + k0 + 32, K, base0 + (kb ^ 1) * 1024, l);
      asm volatile("s_wait_asynccnt 0x2" ::: "memory");
    } else {
      asm volatile("s_wait_asynccnt 0x0" ::: "memory");
    }
    v16h a = load_A(atile0 + kb * 512, 32);
#pragma unroll
    for (int c = 0; c < 4; ++c)
      acc[c] = wmma16(a, load_Bt(W, K, n0 + 16 * c, k0), acc[c]);
  }
#pragma unroll
  for (int c = 0; c < 4; ++c) {
    float bv = bias ? bias[n0 + 16 * c + (l & 15)] : 0.0f;
#pragma unroll
    for (int v = 0; v < 8; ++v) acc[c][v] += bv;
    store_C(Y + (size_t)m0 * N + n0 + 16 * c, N, acc[c]);
  }
}

// ---------------------------------------------------------------- pre-BiLSTM scan
// xg: [T,64,512] f32 (x@Wih^T + b, gates i,f,g,o). Whh: [512,128] f16.
// Quirk: cell h discarded; c masked; h = tanh(c). Writes dir half of Hout f16.
__global__ void k_pre_scan(const float* __restrict__ xg, const f16* __restrict__ Whh,
                           const float* __restrict__ mask, f16* __restrict__ Hout,
                           int T) {
  extern __shared__ char smem[];
  f16* hbuf = (f16*)smem;                                    // [2][64][128]
  float* cbuf = (float*)(smem + 2 * 64 * 128 * sizeof(f16)); // [64][128]
  int dir = blockIdx.x, tid = threadIdx.x, wv = tid >> 5, l = tid & 31;
  int j0 = wv << 4;
  for (int i = tid; i < 64 * 128; i += blockDim.x) {
    hbuf[i] = (f16)0.f; hbuf[64 * 128 + i] = (f16)0.f; cbuf[i] = 0.f;
  }
  // Loop-invariant recurrent weights live in registers for all 600 steps.
  v16h bhh[4][4];
#pragma unroll
  for (int kk = 0; kk < 4; ++kk)
#pragma unroll
    for (int g = 0; g < 4; ++g)
      bhh[kk][g] = load_Bt(Whh, 128, g * 128 + j0, kk * 32);
  __syncthreads();
  for (int s = 0; s < T; ++s) {
    int t = dir ? (T - 1 - s) : s;
    const f16* hc = hbuf + (s & 1) * 64 * 128;
    f16* hn = hbuf + ((s + 1) & 1) * 64 * 128;
    const float* xgt = xg + (size_t)t * 64 * 512;
#pragma unroll
    for (int task = 0; task < 4; ++task) {
      int m0 = task << 4;          // batch tile
      v8f acc[4];
#pragma unroll
      for (int g = 0; g < 4; ++g)
        acc[g] = load_C(xgt + m0 * 512 + g * 128 + j0, 512);
#pragma unroll
      for (int kk = 0; kk < 4; ++kk) {
        v16h a = load_A(hc + m0 * 128 + kk * 32, 128);
#pragma unroll
        for (int g = 0; g < 4; ++g)
          acc[g] = wmma16(a, bhh[kk][g], acc[g]);
      }
      int n = l & 15, mb = (l >> 4) ? 8 : 0;
#pragma unroll
      for (int v = 0; v < 8; ++v) {
        int brow = m0 + mb + v, j = j0 + n;
        float cp = cbuf[brow * 128 + j];
        float iv = fsig(acc[0][v]), fv = fsig(acc[1][v]), gv = ftanh(acc[2][v]);
        float c2 = fv * cp + iv * gv;
        c2 *= mask[(size_t)t * 64 + brow];
        float h2 = ftanh(c2);                       // quirk: o-gate discarded
        cbuf[brow * 128 + j] = c2;
        hn[brow * 128 + j] = (f16)h2;
        Hout[((size_t)t * 64 + brow) * 256 + dir * 128 + j] = (f16)h2;
      }
    }
    __syncthreads();
  }
}

// ---------------------------------------------------------------- Match-LSTM scan
__global__ void k_match_scan(
    const float* __restrict__ aq,  // [60,64,256]  Hq@Wq^T
    const float* __restrict__ ap,  // [600,64,256] Hp@Wp^T + bp
    const f16* __restrict__ Hp,    // [600,64,256] f16
    const f16* __restrict__ Hq,    // [60,64,256]  f16
    const float* __restrict__ Wa,  // [256]
    const float* __restrict__ ba_p,// [1]
    const f16* __restrict__ Wih,   // [512,512]
    const f16* __restrict__ Whh,   // [512,128]
    const f16* __restrict__ Wh,    // [256,128]
    const float* __restrict__ bias,// [512]
    const float* __restrict__ mask,// [600,64]
    float* __restrict__ out,       // [600,64,256] f32
    int T, int TQn) {
  extern __shared__ char smem[];
  float* hwh = (float*)smem;                     // [64][256] f32
  f16* zb    = (f16*)(hwh + 64 * 256);           // [64][512] f16
  f16* hbuf  = zb + 64 * 512;                    // [2][64][128] f16
  float* cbuf = (float*)(hbuf + 2 * 64 * 128);   // [64][128] f32
  float* sbuf = cbuf + 64 * 128;                 // [TQn*64] f32
  float* wal  = sbuf + TQn * 64;                 // [256] f32
  int dir = blockIdx.x, tid = threadIdx.x, wv = tid >> 5, l = tid & 31;
  int j0 = wv << 4;
  float ba = ba_p[0];
  for (int i = tid; i < 64 * 128; i += blockDim.x) {
    hbuf[i] = (f16)0.f; hbuf[64 * 128 + i] = (f16)0.f; cbuf[i] = 0.f;
  }
  for (int i = tid; i < 256; i += blockDim.x) wal[i] = Wa[i];
  // Loop-invariant recurrent weights in registers.
  v16h bhh[4][4];       // Whh frags for this wave's j-tile (4 gates x 4 K-chunks)
#pragma unroll
  for (int kk = 0; kk < 4; ++kk)
#pragma unroll
    for (int g = 0; g < 4; ++g)
      bhh[kk][g] = load_Bt(Whh, 128, g * 128 + j0, kk * 32);
  v16h bwh[2][4];       // Wh frags: this wave touches n-tiles wv and wv+8 only
#pragma unroll
  for (int half = 0; half < 2; ++half)
#pragma unroll
    for (int kk = 0; kk < 4; ++kk)
      bwh[half][kk] = load_Bt(Wh, 128, (wv + 8 * half) * 16, kk * 32);
  __syncthreads();
  for (int s = 0; s < T; ++s) {
    int t = dir ? (T - 1 - s) : s;
    const f16* hc = hbuf + (s & 1) * 64 * 128;
    f16* hn = hbuf + ((s + 1) & 1) * 64 * 128;
    // 1) hwh = h @ Wh^T : [64,128] x [128,256]
#pragma unroll
    for (int it = 0; it < 8; ++it) {
      int f = wv + it * 8;
      int m0 = (f >> 4) << 4, n0 = (f & 15) << 4;
      v8f acc = {};
#pragma unroll
      for (int kk = 0; kk < 4; ++kk) {
        v16h a = load_A(hc + m0 * 128 + kk * 32, 128);
        acc = wmma16(a, bwh[it & 1][kk], acc);
      }
      store_C(hwh + m0 * 256 + n0, 256, acc);
    }
    __syncthreads();
    // 2) attention scores s[tq,b] = sum_c tanh(aq+ap+hwh)*Wa + ba
    const float* apt = ap + (size_t)t * 64 * 256;
    for (int p = tid; p < TQn * 64; p += blockDim.x) {
      int tq = p >> 6, b = p & 63;
      const float4* aqr = (const float4*)(aq + ((size_t)tq * 64 + b) * 256);
      const float4* apr = (const float4*)(apt + (size_t)b * 256);
      const float4* hwr = (const float4*)(hwh + b * 256);
      const float4* wa4 = (const float4*)wal;
      float sum = 0.f;
      for (int c = 0; c < 64; ++c) {
        float4 xa = aqr[c], xp = apr[c], xh = hwr[c], xw = wa4[c];
        sum += ftanh(xa.x + xp.x + xh.x) * xw.x;
        sum += ftanh(xa.y + xp.y + xh.y) * xw.y;
        sum += ftanh(xa.z + xp.z + xh.z) * xw.z;
        sum += ftanh(xa.w + xp.w + xh.w) * xw.w;
      }
      sbuf[p] = sum + ba;
    }
    __syncthreads();
    // 3) softmax over tq per batch column
    if (tid < 64) {
      float mx = -1e30f;
      for (int tq = 0; tq < TQn; ++tq) mx = fmaxf(mx, sbuf[tq * 64 + tid]);
      float sm = 0.f;
      for (int tq = 0; tq < TQn; ++tq) {
        float e = __expf(sbuf[tq * 64 + tid] - mx);
        sbuf[tq * 64 + tid] = e; sm += e;
      }
      float inv = 1.f / sm;
      for (int tq = 0; tq < TQn; ++tq) sbuf[tq * 64 + tid] *= inv;
    }
    __syncthreads();
    // 4) z = [Hp[t], sum_tq alpha*Hq]
    const f16* hpt = Hp + (size_t)t * 64 * 256;
    for (int o = tid; o < 64 * 256; o += blockDim.x) {
      int b = o >> 8, c = o & 255;
      float acc = 0.f;
      for (int tq = 0; tq < TQn; ++tq)
        acc += sbuf[tq * 64 + b] * (float)Hq[((size_t)tq * 64 + b) * 256 + c];
      zb[b * 512 + 256 + c] = (f16)acc;
      zb[b * 512 + c] = hpt[b * 256 + c];
    }
    __syncthreads();
    // 5) gates = z@Wih^T + h@Whh^T + b ; full LSTM cell; mask h and c
#pragma unroll
    for (int task = 0; task < 4; ++task) {
      int m0 = task << 4;
      int n = l & 15, mb = (l >> 4) ? 8 : 0;
      v8f acc[4];
#pragma unroll
      for (int g = 0; g < 4; ++g) {
        float bv = bias[g * 128 + j0 + n];
        v8f a;
#pragma unroll
        for (int v = 0; v < 8; ++v) a[v] = bv;
        acc[g] = a;
      }
      for (int k0 = 0; k0 < 512; k0 += 32) {
        v16h a = load_A(zb + m0 * 512 + k0, 512);
#pragma unroll
        for (int g = 0; g < 4; ++g)
          acc[g] = wmma16(a, load_Bt(Wih, 512, g * 128 + j0, k0), acc[g]);
      }
#pragma unroll
      for (int kk = 0; kk < 4; ++kk) {
        v16h a = load_A(hc + m0 * 128 + kk * 32, 128);
#pragma unroll
        for (int g = 0; g < 4; ++g)
          acc[g] = wmma16(a, bhh[kk][g], acc[g]);
      }
#pragma unroll
      for (int v = 0; v < 8; ++v) {
        int brow = m0 + mb + v, j = j0 + n;
        float cp = cbuf[brow * 128 + j];
        float iv = fsig(acc[0][v]), fv = fsig(acc[1][v]);
        float gv = ftanh(acc[2][v]), ov = fsig(acc[3][v]);
        float ct = fv * cp + iv * gv;
        float m = mask[(size_t)t * 64 + brow];
        float h2 = ov * ftanh(ct) * m;
        float c2 = ct * m;
        cbuf[brow * 128 + j] = c2;
        hn[brow * 128 + j] = (f16)h2;
        out[((size_t)t * 64 + brow) * 256 + dir * 128 + j] = h2;
      }
    }
    __syncthreads();
  }
}

// ---------------------------------------------------------------- host
extern "C" void kernel_launch(void* const* d_in, const int* in_sizes, int n_in,
                              void* d_out, int out_size, void* d_ws, size_t ws_size,
                              hipStream_t stream) {
  const float* passage  = (const float*)d_in[0];
  const float* question = (const float*)d_in[1];
  const float* mask_p   = (const float*)d_in[2];
  const float* mask_q   = (const float*)d_in[3];
  const float* pre0_Wih = (const float*)d_in[4];
  const float* pre0_Whh = (const float*)d_in[5];
  const float* pre0_b   = (const float*)d_in[6];
  const float* pre1_Wih = (const float*)d_in[7];
  const float* pre1_Whh = (const float*)d_in[8];
  const float* pre1_b   = (const float*)d_in[9];
  const float* mq_Wq  = (const float*)d_in[10];
  const float* mq_Wp  = (const float*)d_in[11];
  const float* mq_bp  = (const float*)d_in[12];
  const float* mq_Wh  = (const float*)d_in[13];
  const float* mq_Wa  = (const float*)d_in[14];
  const float* mq_ba  = (const float*)d_in[15];
  const float* mq_Wih = (const float*)d_in[16];
  const float* mq_Whh = (const float*)d_in[17];
  const float* mq_b   = (const float*)d_in[18];
  float* out = (float*)d_out;

  char* w = (char*)d_ws;
  auto alloc = [&](size_t bytes) {
    char* p = w; w += (bytes + 255) & ~(size_t)255; return p;
  };
  const size_t MP = (size_t)TPn * Bn, MQ = (size_t)TQN * Bn;
  f16* P_h    = (f16*)alloc(MP * EMBP * 2);
  f16* Q_h    = (f16*)alloc(MQ * EMBP * 2);
  f16* W0ih_h = (f16*)alloc(512 * EMBP * 2);
  f16* W0hh_h = (f16*)alloc(512 * 128 * 2);
  f16* W1ih_h = (f16*)alloc(512 * 256 * 2);
  f16* W1hh_h = (f16*)alloc(512 * 128 * 2);
  f16* Wq_h   = (f16*)alloc(256 * 256 * 2);
  f16* Wp_h   = (f16*)alloc(256 * 256 * 2);
  f16* Wh_h   = (f16*)alloc(256 * 128 * 2);
  f16* Wih_h  = (f16*)alloc(512 * 512 * 2);
  f16* Whh_h  = (f16*)alloc(512 * 128 * 2);
  float* xg_p = (float*)alloc(MP * 512 * 4);
  float* xg_q = (float*)alloc(MQ * 512 * 4);
  f16* Hp0 = (f16*)alloc(MP * 256 * 2);
  f16* Hq0 = (f16*)alloc(MQ * 256 * 2);
  f16* Hp1 = (f16*)alloc(MP * 256 * 2);
  f16* Hq1 = (f16*)alloc(MQ * 256 * 2);
  float* aq = (float*)alloc(MQ * 256 * 4);
  float* ap = (float*)alloc(MP * 256 * 4);

  auto cvt = [&](const float* src, f16* dst, long rows, int K, int KP) {
    long tot = rows * KP;
    k_cvt_pad<<<dim3((unsigned)((tot + 255) / 256)), dim3(256), 0, stream>>>(
        src, dst, rows, K, KP);
  };
  cvt(passage,  P_h,    (long)MP, EMB, EMBP);
  cvt(question, Q_h,    (long)MQ, EMB, EMBP);
  cvt(pre0_Wih, W0ih_h, 512, EMB, EMBP);
  cvt(pre0_Whh, W0hh_h, 512, 128, 128);
  cvt(pre1_Wih, W1ih_h, 512, 256, 256);
  cvt(pre1_Whh, W1hh_h, 512, 128, 128);
  cvt(mq_Wq,  Wq_h,  256, 256, 256);
  cvt(mq_Wp,  Wp_h,  256, 256, 256);
  cvt(mq_Wh,  Wh_h,  256, 128, 128);
  cvt(mq_Wih, Wih_h, 512, 512, 512);
  cvt(mq_Whh, Whh_h, 512, 128, 128);

  const size_t lds_gemm = 8 * 2048;  // 8 waves x double-buffered 1KB A tile
  auto gemm = [&](const f16* X, const f16* W_, const float* b, float* Y,
                  int M, int N, int K) {
    int blocks = (M / 16) * (N / 64) / 8;   // 8 waves per 256-thread block
    k_gemm<<<dim3(blocks), dim3(256), lds_gemm, stream>>>(X, W_, b, Y, M, N, K);
  };
  const size_t lds_pre = 2 * 64 * 128 * sizeof(f16) + 64 * 128 * sizeof(float);
  const size_t lds_match = 64 * 256 * 4 + 64 * 512 * 2 + 2 * 64 * 128 * 2 +
                           64 * 128 * 4 + (size_t)TQN * 64 * 4 + 256 * 4;

  // --- pre layer 0 ---
  gemm(P_h, W0ih_h, pre0_b, xg_p, (int)MP, 512, EMBP);
  k_pre_scan<<<dim3(2), dim3(256), lds_pre, stream>>>(xg_p, W0hh_h, mask_p, Hp0, TPn);
  gemm(Q_h, W0ih_h, pre0_b, xg_q, (int)MQ, 512, EMBP);
  k_pre_scan<<<dim3(2), dim3(256), lds_pre, stream>>>(xg_q, W0hh_h, mask_q, Hq0, TQN);
  // --- pre layer 1 ---
  gemm(Hp0, W1ih_h, pre1_b, xg_p, (int)MP, 512, 256);
  k_pre_scan<<<dim3(2), dim3(256), lds_pre, stream>>>(xg_p, W1hh_h, mask_p, Hp1, TPn);
  gemm(Hq0, W1ih_h, pre1_b, xg_q, (int)MQ, 512, 256);
  k_pre_scan<<<dim3(2), dim3(256), lds_pre, stream>>>(xg_q, W1hh_h, mask_q, Hq1, TQN);
  // --- match precomputes ---
  gemm(Hq1, Wq_h, nullptr, aq, (int)MQ, 256, 256);
  gemm(Hp1, Wp_h, mq_bp,   ap, (int)MP, 256, 256);
  // --- match scan ---
  k_match_scan<<<dim3(2), dim3(256), lds_match, stream>>>(
      aq, ap, Hp1, Hq1, mq_Wa, mq_ba, Wih_h, Whh_h, Wh_h, mq_b, mask_p,
      out, TPn, TQN);
  (void)in_sizes; (void)n_in; (void)out_size; (void)ws_size;
}
